// DisentangledMaskAttention_78082505441290
// MI455X (gfx1250) — matrix operand order repaired and checked by reading, not compiled, not run on hardware
//
#include <hip/hip_runtime.h>
#include <hip/hip_bf16.h>
#include <math.h>

// Problem constants (from reference setup): B=4, L=1024, D=512, H=8, C=4, dk=64
#define B_ 4
#define L_ 1024
#define D_ 512
#define H_ 8
#define C_ 4
#define DK_ 64
#define LOG_VAR0 (-4.6051702f)   // log(0.01)

typedef __bf16 v16bf __attribute__((ext_vector_type(16)));
typedef __bf16 v8bf  __attribute__((ext_vector_type(8)));
typedef float  v8f   __attribute__((ext_vector_type(8)));

// A-fragment (16x32 bf16, MxK) layout used here (k contiguous per lane half, same order as lds16):
// lane m=lane&15, kh=lane>>4; VGPR v (pair of halves): k = 2v + 16*kh
static __device__ __forceinline__ v16bf ldsA(const __bf16* base, int stride, int lane, int koff) {
  int m = lane & 15, kh = lane >> 4;
  const __bf16* p = base + m * stride + koff;
  v16bf a;
#pragma unroll
  for (int v = 0; v < 8; ++v) {
    int k = 2 * v + 16 * kh;
    a[2 * v]     = p[k];
    a[2 * v + 1] = p[k + 1];
  }
  return a;
}

// B-fragment (32x16 bf16, KxN): lane n=lane&15 holds 16 contiguous K values,
// lanes 0-15 -> K 0..15, lanes 16-31 -> K 16..31 (caller passes the right slice ptr)
static __device__ __forceinline__ v16bf lds16(const __bf16* p) {
  v16bf b;
#pragma unroll
  for (int j = 0; j < 16; ++j) b[j] = p[j];
  return b;
}

static __device__ __forceinline__ v8f wmma_bf16(v16bf a, v16bf b, v8f c) {
  return __builtin_amdgcn_wmma_f32_16x16x32_bf16(false, a, false, b, (short)0, c, false, false);
}

static __device__ __forceinline__ float rmax16(float v) {
#pragma unroll
  for (int m = 1; m < 16; m <<= 1) v = fmaxf(v, __shfl_xor(v, m, 32));
  return v;
}
static __device__ __forceinline__ float rsum16(float v) {
#pragma unroll
  for (int m = 1; m < 16; m <<= 1) v += __shfl_xor(v, m, 32);
  return v;
}

// widened f32x8 -> bf16x8 convert (pairs fuse into v_cvt_pk_bf16_f32)
static __device__ __forceinline__ v8bf cvt8(const float* __restrict__ p) {
  float4 a = *(const float4*)p;
  float4 b = *(const float4*)(p + 4);
  v8bf r;
  r[0] = (__bf16)a.x; r[1] = (__bf16)a.y; r[2] = (__bf16)a.z; r[3] = (__bf16)a.w;
  r[4] = (__bf16)b.x; r[5] = (__bf16)b.y; r[6] = (__bf16)b.z; r[7] = (__bf16)b.w;
  return r;
}

// ---- CDNA5 async global->LDS copy (ASYNCcnt-tracked, no VGPR round trip) ----
static __device__ __forceinline__ unsigned lds_addr_of(const void* p) {
  return (unsigned)(unsigned long long)(__attribute__((address_space(3))) const char*)p;
}
static __device__ __forceinline__ void async_ld16(unsigned lds, const void* g) {
  asm volatile("global_load_async_to_lds_b128 %0, %1, off"
               :: "v"(lds), "v"(g) : "memory");
}
static __device__ __forceinline__ void async_wait0() {
  asm volatile("s_wait_asynccnt 0x0" ::: "memory");
}

// ---------------------------------------------------------------------------
// GEMM: out[m,n] = sum_k A[m,k]*W[n,k] + bias[n]   (NT, M=4096, N=512, K=512)
// mode 0: row-major [M,N] f32 out.
// mode 1: f32 scatter to [B,H,L,DK] + bf16 copy (same layout).
// mode 2: bf16 transposed scatter to [B,H,DK,L] only (for V).
// ---------------------------------------------------------------------------
__global__ __launch_bounds__(128) void gemm_nt_kernel(
    const float* __restrict__ A, const float* __restrict__ W,
    const float* __restrict__ bias, float* __restrict__ outf,
    __bf16* __restrict__ outb, int mode) {
  __shared__ __align__(16) __bf16 As[64][32];
  __shared__ __align__(16) __bf16 Bs[64][32];
  const int bm = blockIdx.x * 64, bn = blockIdx.y * 64;
  const int tid = threadIdx.x, lane = tid & 31, wv = tid >> 5;
  const int n16 = lane & 15, kh = lane >> 4;
  v8f acc[4] = {};
  for (int k0 = 0; k0 < D_; k0 += 32) {
    __syncthreads();
    // 256 chunks of 8 elements: global_load_b128 x2 -> cvt_pk -> ds_store_b128
    for (int i = tid; i < (64 * 32) / 8; i += 128) {
      int r = i >> 2, cc = (i & 3) * 8;
      *(v8bf*)&As[r][cc] = cvt8(&A[(size_t)(bm + r) * D_ + k0 + cc]);
      *(v8bf*)&Bs[r][cc] = cvt8(&W[(size_t)(bn + r) * D_ + k0 + cc]);
    }
    if (k0 + 32 < D_) {
      __builtin_prefetch(&A[(size_t)(bm + (tid >> 1)) * D_ + k0 + 32], 0, 1);
      __builtin_prefetch(&W[(size_t)(bn + (tid >> 1)) * D_ + k0 + 32], 0, 1);
    }
    __syncthreads();
    v16bf a = ldsA(&As[wv * 16][0], 32, lane, 0);
#pragma unroll
    for (int j = 0; j < 4; ++j) {
      v16bf bfr = lds16(&Bs[j * 16 + n16][kh * 16]);
      acc[j] = wmma_bf16(a, bfr, acc[j]);
    }
  }
#pragma unroll
  for (int j = 0; j < 4; ++j) {
#pragma unroll
    for (int r = 0; r < 8; ++r) {
      int m = bm + wv * 16 + r + 8 * kh;
      int n = bn + j * 16 + n16;
      float v = acc[j][r] + bias[n];
      int b = m >> 10, l = m & (L_ - 1);
      int hh = n >> 6, d = n & (DK_ - 1);
      if (mode == 0) {
        outf[(size_t)m * D_ + n] = v;
      } else if (mode == 1) {
        size_t dst = (((size_t)b * H_ + hh) * L_ + l) * DK_ + d;
        outf[dst] = v;
        outb[dst] = (__bf16)v;
      } else {  // mode 2: transposed bf16 [B,H,DK,L]
        outb[(((size_t)b * H_ + hh) * DK_ + d) * L_ + l] = (__bf16)v;
      }
    }
  }
}

// ---------------------------------------------------------------------------
// Clustering: per (b,h) computes cp[B,H,L,C], accumulates KL and div (Gram trick)
// ---------------------------------------------------------------------------
__global__ __launch_bounds__(256) void cluster_kernel(
    const float* __restrict__ x,        // [B,H,L,DK] f32
    const float* __restrict__ mu,       // [H,C,DK]
    const float* __restrict__ logvar,   // [H,C,DK]
    const float* __restrict__ logprior, // [H,C]
    float* __restrict__ cp,             // [B,H,L,C]
    float* __restrict__ klb, float* __restrict__ dvb) {
  const int bh = blockIdx.x, b = bh / H_, h = bh % H_;
  const int tid = threadIdx.x;
  __shared__ float mu_s[C_][DK_], iv_s[C_][DK_];
  __shared__ float lvsum[C_], ivsum[C_], lp_s[C_], prior_s[C_];
  __shared__ float red[12];   // [0]=kl, [1..10]=sym M, [11]=diag term

  {
    int c = tid >> 6, d = tid & 63;           // 256 threads exactly cover C*DK
    float lv = logvar[((size_t)h * C_ + c) * DK_ + d];
    mu_s[c][d] = mu[((size_t)h * C_ + c) * DK_ + d];
    iv_s[c][d] = __expf(-lv);
  }
  if (tid < 12) red[tid] = 0.f;
  if (tid < C_) {
    float s1 = 0.f, s2 = 0.f;
    for (int d = 0; d < DK_; ++d) {
      float lv = logvar[((size_t)h * C_ + tid) * DK_ + d];
      s1 += lv; s2 += __expf(-lv);
    }
    lvsum[tid] = s1; ivsum[tid] = s2;
  }
  if (tid == 0) {
    float mp = -1e30f;
    for (int c = 0; c < C_; ++c) mp = fmaxf(mp, logprior[h * C_ + c]);
    float se = 0.f;
    for (int c = 0; c < C_; ++c) se += __expf(logprior[h * C_ + c] - mp);
    float lse = mp + __logf(se);
    for (int c = 0; c < C_; ++c) {
      lp_s[c] = logprior[h * C_ + c] - lse;
      prior_s[c] = __expf(lp_s[c]);
    }
  }
  __syncthreads();

  float kl_loc = 0.f, dloc = 0.f, Mloc[10];
#pragma unroll
  for (int i = 0; i < 10; ++i) Mloc[i] = 0.f;

  for (int l = tid; l < L_; l += 256) {
    const float* xr = x + ((size_t)bh * L_ + l) * DK_;
    float ms[C_] = {0.f, 0.f, 0.f, 0.f};
    for (int d = 0; d < DK_; d += 4) {
      float4 xv4 = *(const float4*)&xr[d];
      float xv[4] = {xv4.x, xv4.y, xv4.z, xv4.w};
#pragma unroll
      for (int u = 0; u < 4; ++u)
#pragma unroll
        for (int c = 0; c < C_; ++c) {
          float df = xv[u] - mu_s[c][d + u];
          ms[c] += df * df * iv_s[c][d + u];
        }
    }
    float lpdf[C_], mx = -1e30f;
#pragma unroll
    for (int c = 0; c < C_; ++c) {
      lpdf[c] = -0.5f * ms[c] - 0.5f * lvsum[c] + lp_s[c];
      mx = fmaxf(mx, lpdf[c]);
    }
    float e[C_], se = 0.f;
#pragma unroll
    for (int c = 0; c < C_; ++c) { e[c] = __expf(lpdf[c] - mx); se += e[c]; }
    float inv = 1.f / se, lse = mx + __logf(se);
    float p[C_];
    float* cpr = cp + ((size_t)bh * L_ + l) * C_;
    float g = 0.f;
#pragma unroll
    for (int c = 0; c < C_; ++c) {
      p[c] = e[c] * inv;
      cpr[c] = p[c];
      g += p[c] * p[c];
      kl_loc += prior_s[c] * (lp_s[c] - (lpdf[c] - lse));
      kl_loc += 0.5f * p[c] * (ms[c] + 0.01f * ivsum[c] + lvsum[c]);
    }
    dloc += 1.25f * (g - 1.f) * (g - 1.f) - 0.75f * g * g;
    int idx = 0;
#pragma unroll
    for (int c = 0; c < C_; ++c)
#pragma unroll
      for (int c2 = c; c2 < C_; ++c2) Mloc[idx++] += p[c] * p[c2];
  }

  atomicAdd(&red[0], kl_loc);
#pragma unroll
  for (int i = 0; i < 10; ++i) atomicAdd(&red[1 + i], Mloc[i]);
  atomicAdd(&red[11], dloc);
  __syncthreads();
  if (tid == 0) {
    float frob = 0.f;
    int idx = 0;
    for (int c = 0; c < C_; ++c)
      for (int c2 = c; c2 < C_; ++c2) {
        float v = red[1 + idx++];
        frob += (c == c2) ? v * v : 2.f * v * v;
      }
    atomicAdd(klb + b, red[0] / (float)(H_ * L_));
    atomicAdd(dvb + b, (0.75f * frob + red[11]) / ((float)H_ * (float)L_ * (float)L_));
  }
}

// ---------------------------------------------------------------------------
// Mean over L of q -> xh[B*H, DK]
// ---------------------------------------------------------------------------
__global__ __launch_bounds__(64) void qmean_kernel(const float* __restrict__ q,
                                                   float* __restrict__ xh) {
  int bh = blockIdx.x, d = threadIdx.x;
  float s = 0.f;
  for (int l = 0; l < L_; ++l) s += q[((size_t)bh * L_ + l) * DK_ + d];
  xh[(size_t)bh * DK_ + d] = s * (1.f / (float)L_);
}

// ---------------------------------------------------------------------------
// Head clustering on xh (tokens = H_ per batch), adds to kl/div
// ---------------------------------------------------------------------------
__global__ __launch_bounds__(32) void headcluster_kernel(
    const float* __restrict__ xh, const float* __restrict__ mu,
    const float* __restrict__ lv, const float* __restrict__ lprior,
    float* __restrict__ klb, float* __restrict__ dvb) {
  int t = threadIdx.x;           // 32 threads = B_*H_ tokens
  int b = t >> 3, i = t & 7;
  float mp = -1e30f;
  for (int c = 0; c < C_; ++c) mp = fmaxf(mp, lprior[c]);
  float se0 = 0.f;
  for (int c = 0; c < C_; ++c) se0 += __expf(lprior[c] - mp);
  float lseP = mp + __logf(se0);
  float lp[C_], lvs[C_], ivs[C_], ms[C_];
  const float* xr = xh + (size_t)t * DK_;
  for (int c = 0; c < C_; ++c) {
    lp[c] = lprior[c] - lseP;
    float s1 = 0.f, s2 = 0.f, m2 = 0.f;
    for (int d = 0; d < DK_; ++d) {
      float l_ = lv[c * DK_ + d];
      float ivd = __expf(-l_);
      s1 += l_; s2 += ivd;
      float df = xr[d] - mu[c * DK_ + d];
      m2 += df * df * ivd;
    }
    lvs[c] = s1; ivs[c] = s2; ms[c] = m2;
  }
  float lpdf[C_], mx = -1e30f;
  for (int c = 0; c < C_; ++c) {
    lpdf[c] = -0.5f * ms[c] - 0.5f * lvs[c] + lp[c];
    mx = fmaxf(mx, lpdf[c]);
  }
  float e[C_], se = 0.f;
  for (int c = 0; c < C_; ++c) { e[c] = __expf(lpdf[c] - mx); se += e[c]; }
  float lse = mx + __logf(se), inv = 1.f / se;
  __shared__ float cps[32][C_];
  __shared__ float kls[32];
  float klt = 0.f;
  for (int c = 0; c < C_; ++c) {
    float p = e[c] * inv;
    cps[t][c] = p;
    klt += __expf(lp[c]) * (lp[c] - (lpdf[c] - lse));
    klt += 0.5f * p * (ms[c] + 0.01f * ivs[c] + lvs[c]);
  }
  kls[t] = klt;
  __syncthreads();
  if (i == 0) {
    float klsum = 0.f;
    for (int ii = 0; ii < 8; ++ii) klsum += kls[b * 8 + ii];
    float dv = 0.f;
    for (int qq = 0; qq < 8; ++qq)
      for (int kk = 0; kk < 8; ++kk) {
        float g = 0.f;
        for (int c = 0; c < C_; ++c) g += cps[b * 8 + qq][c] * cps[b * 8 + kk][c];
        float eye = (qq == kk) ? 1.f : 0.f;
        float w = (qq == kk) ? 1.25f : 0.75f;
        dv += (g - eye) * (g - eye) * w;
      }
    atomicAdd(klb + b, klsum * (1.f / 8.f));
    atomicAdd(dvb + b, dv * (1.f / 64.f));
  }
}

// ---------------------------------------------------------------------------
// Fused flash attention with cluster mask + renormalization.
// Block = 128 thr (4 waves), 64 q-rows of one (b,h); k-tile = 32 keys.
// Tile staging uses CDNA5 async global->LDS copies (bf16 data, pre-converted).
// ---------------------------------------------------------------------------
__global__ __launch_bounds__(128) void attn_kernel(
    const __bf16* __restrict__ qb,   // [B,H,L,DK] bf16
    const __bf16* __restrict__ kb,   // [B,H,L,DK] bf16
    const __bf16* __restrict__ vT,   // [B,H,DK,L] bf16 (pre-transposed)
    const float* __restrict__ pqg, const float* __restrict__ pkg,
    const int* __restrict__ pmask,
    float* __restrict__ ctx) {       // ctx written as [B, L, H*DK]
  const int bh = blockIdx.y, b = bh / H_, h = bh % H_;
  const int qt = blockIdx.x;
  const int tid = threadIdx.x, lane = tid & 31, wv = tid >> 5;
  const int n16 = lane & 15, kh = lane >> 4;

  __shared__ __align__(16) __bf16 Qs[64][64];
  __shared__ __align__(16) __bf16 Ks[32][64];
  __shared__ __align__(16) __bf16 Vt[64][32];   // V transposed: [d][key]
  __shared__ __align__(16) float  pq_s[64][C_];
  __shared__ __align__(16) float  pk_s[32][C_];
  __shared__ float  pm_s[32];
  __shared__ __align__(16) __bf16 Ps[4][16][32];  // per-wave P staging

  const unsigned qs_off = lds_addr_of(&Qs[0][0]);
  const unsigned ks_off = lds_addr_of(&Ks[0][0]);
  const unsigned vt_off = lds_addr_of(&Vt[0][0]);

  // stage Q tile (contiguous 64*64 bf16 = 8KB) via async copies
  const __bf16* qbase = qb + ((size_t)bh * L_ + qt * 64) * DK_;
  for (int i = tid; i < (64 * 64) / 8; i += 128)
    async_ld16(qs_off + i * 16, (const void*)(qbase + i * 8));
  for (int i = tid; i < 64; i += 128)
    *(float4*)&pq_s[i][0] = *(const float4*)&pqg[((size_t)bh * L_ + qt * 64 + i) * C_];
  async_wait0();
  __syncthreads();

  const v16bf aq0 = ldsA(&Qs[wv * 16][0], 64, lane, 0);
  const v16bf aq1 = ldsA(&Qs[wv * 16][0], 64, lane, 32);

  float Mx[8], Sw[8], Zs[8];
#pragma unroll
  for (int r = 0; r < 8; ++r) { Mx[r] = -1e30f; Sw[r] = 0.f; Zs[r] = 0.f; }
  v8f o0 = {}, o1 = {}, o2 = {}, o3 = {};

  for (int kt = 0; kt < L_ / 32; ++kt) {
    __syncthreads();
    // K tile: contiguous 32*64 bf16 = 4KB
    const __bf16* kbase = kb + ((size_t)bh * L_ + kt * 32) * DK_;
    for (int i = tid; i < (32 * 64) / 8; i += 128)
      async_ld16(ks_off + i * 16, (const void*)(kbase + i * 8));
    // Vt tile: 64 rows (d) x 32 keys, each row 4 x 16B chunks from [B,H,DK,L]
    for (int i = tid; i < 64 * 4; i += 128) {
      int d = i >> 2, part = i & 3;
      const __bf16* src = vT + (((size_t)bh * DK_ + d) * L_ + kt * 32 + part * 8);
      async_ld16(vt_off + (unsigned)(d * 32 + part * 8) * 2, (const void*)src);
    }
    if (tid < 32) {
      *(float4*)&pk_s[tid][0] = *(const float4*)&pkg[((size_t)bh * L_ + kt * 32 + tid) * C_];
      pm_s[tid] = (pmask[(size_t)b * L_ + kt * 32 + tid] == 0) ? 0.f : 1.f;
    }
    async_wait0();
    __syncthreads();

    // scores: S = Q (16x64) . K^T (64x32), two 16-col halves, K-split 32+32
    v8f s0 = {}, s1 = {};
    s0 = wmma_bf16(aq0, lds16(&Ks[n16][kh * 16]), s0);
    s0 = wmma_bf16(aq1, lds16(&Ks[n16][32 + kh * 16]), s0);
    s1 = wmma_bf16(aq0, lds16(&Ks[16 + n16][kh * 16]), s1);
    s1 = wmma_bf16(aq1, lds16(&Ks[16 + n16][32 + kh * 16]), s1);

    float pk0[C_], pk1[C_];
#pragma unroll
    for (int c = 0; c < C_; ++c) { pk0[c] = pk_s[n16][c]; pk1[c] = pk_s[16 + n16][c]; }
    const float pm0 = pm_s[n16], pm1 = pm_s[16 + n16];

#pragma unroll
    for (int r = 0; r < 8; ++r) {
      int row = wv * 16 + r + 8 * kh;
      float cm0 = 0.f, cm1 = 0.f;
#pragma unroll
      for (int c = 0; c < C_; ++c) {
        float pqv = pq_s[row][c];
        cm0 += pqv * pk0[c];
        cm1 += pqv * pk1[c];
      }
      float sc0 = (pm0 != 0.f) ? s0[r] * 0.125f : -1e30f;
      float sc1 = (pm1 != 0.f) ? s1[r] * 0.125f : -1e30f;
      float tmax = rmax16(fmaxf(sc0, sc1));
      float newM = fmaxf(Mx[r], tmax);
      float alpha = __expf(Mx[r] - newM);
      float e0 = __expf(sc0 - newM) * pm0;
      float e1 = __expf(sc1 - newM) * pm1;
      float w0 = e0 * cm0, w1 = e1 * cm1;
      float zsum = rsum16(e0 + e1);
      float ssum = rsum16(w0 + w1);
      Mx[r] = newM;
      Zs[r] = Zs[r] * alpha + zsum;
      Sw[r] = Sw[r] * alpha + ssum;
      o0[r] *= alpha; o1[r] *= alpha; o2[r] *= alpha; o3[r] *= alpha;
      Ps[wv][r + 8 * kh][n16]      = (__bf16)w0;
      Ps[wv][r + 8 * kh][16 + n16] = (__bf16)w1;
    }

    // ctx += P (16x32) . V (32x64); LDS same-wave RAW is in-order per ISA
    v16bf ap = ldsA(&Ps[wv][0][0], 32, lane, 0);
    o0 = wmma_bf16(ap, lds16(&Vt[0 * 16 + n16][kh * 16]), o0);
    o1 = wmma_bf16(ap, lds16(&Vt[1 * 16 + n16][kh * 16]), o1);
    o2 = wmma_bf16(ap, lds16(&Vt[2 * 16 + n16][kh * 16]), o2);
    o3 = wmma_bf16(ap, lds16(&Vt[3 * 16 + n16][kh * 16]), o3);
  }

#pragma unroll
  for (int r = 0; r < 8; ++r) {
    float denom = fmaxf(fmaxf(Sw[r], 1e-6f * Zs[r]), 1e-30f);
    float inv = 1.f / denom;
    int l = qt * 64 + wv * 16 + r + 8 * kh;
    size_t rowoff = ((size_t)b * L_ + l) * D_ + (size_t)h * DK_;
    ctx[rowoff + 0 * 16 + n16] = o0[r] * inv;
    ctx[rowoff + 1 * 16 + n16] = o1[r] * inv;
    ctx[rowoff + 2 * 16 + n16] = o2[r] * inv;
    ctx[rowoff + 3 * 16 + n16] = o3[r] * inv;
  }
}

// ---------------------------------------------------------------------------
__global__ __launch_bounds__(32) void finalize_kernel(const float* __restrict__ klb,
                                                      const float* __restrict__ dvb,
                                                      float* __restrict__ out) {
  int b = threadIdx.x;
  if (b < B_) {
    const size_t base = (size_t)B_ * L_ * D_;
    // -0.5*D*(1+alv).mean term, once per clustering call (q, k, head), alv const
    float klconst = 3.f * (-0.5f * (float)DK_ * (1.f + LOG_VAR0));
    out[base + b]      = klb[b] + klconst;
    out[base + B_ + b] = dvb[b];
  }
}

extern "C" void kernel_launch(void* const* d_in, const int* in_sizes, int n_in,
                              void* d_out, int out_size, void* d_ws, size_t ws_size,
                              hipStream_t stream) {
  const float* query = (const float*)d_in[0];
  const float* key_  = (const float*)d_in[1];
  const float* value = (const float*)d_in[2];
  const int*   pmask = (const int*)d_in[3];
  const float* Wq = (const float*)d_in[4];
  const float* bq = (const float*)d_in[5];
  const float* Wk = (const float*)d_in[6];
  const float* bk = (const float*)d_in[7];
  const float* Wv = (const float*)d_in[8];
  const float* bv = (const float*)d_in[9];
  const float* Wo = (const float*)d_in[10];
  const float* bo = (const float*)d_in[11];
  const float* smu = (const float*)d_in[12];
  const float* slv = (const float*)d_in[13];
  const float* slp = (const float*)d_in[14];
  const float* hmu = (const float*)d_in[15];
  const float* hlv = (const float*)d_in[16];
  const float* hlp = (const float*)d_in[17];

  float* ws = (float*)d_ws;
  const size_t NQ = (size_t)B_ * L_ * D_;          // 2,097,152
  const size_t NCP = (size_t)B_ * H_ * L_ * C_;    // 131,072
  float* q_ws  = ws;                                // f32 q (clustering)
  float* k_ws  = q_ws + NQ;                         // f32 k (clustering)
  float* ctx   = k_ws + NQ;
  float* cpq   = ctx + NQ;
  float* cpk   = cpq + NCP;
  float* xh    = cpk + NCP;                         // B*H*DK = 2048
  float* klb   = xh + (size_t)B_ * H_ * DK_;
  float* dvb   = klb + B_;
  __bf16* qb   = (__bf16*)(dvb + B_);               // bf16 q [B,H,L,DK]
  __bf16* kbb  = qb + NQ;                           // bf16 k [B,H,L,DK]
  __bf16* vTb  = kbb + NQ;                          // bf16 v [B,H,DK,L]

  hipMemsetAsync(klb, 0, 2 * B_ * sizeof(float), stream);

  dim3 gg(64, 8);   // M/64=64, N/64=8
  gemm_nt_kernel<<<gg, 128, 0, stream>>>(query, Wq, bq, q_ws, qb, 1);
  gemm_nt_kernel<<<gg, 128, 0, stream>>>(key_,  Wk, bk, k_ws, kbb, 1);
  gemm_nt_kernel<<<gg, 128, 0, stream>>>(value, Wv, bv, nullptr, vTb, 2);

  cluster_kernel<<<B_ * H_, 256, 0, stream>>>(q_ws, smu, slv, slp, cpq, klb, dvb);
  cluster_kernel<<<B_ * H_, 256, 0, stream>>>(k_ws, smu, slv, slp, cpk, klb, dvb);

  qmean_kernel<<<B_ * H_, 64, 0, stream>>>(q_ws, xh);
  headcluster_kernel<<<1, 32, 0, stream>>>(xh, hmu, hlv, hlp, klb, dvb);

  attn_kernel<<<dim3(L_ / 64, B_ * H_), 128, 0, stream>>>(qb, kbb, vTb, cpq, cpk,
                                                          pmask, ctx);

  gemm_nt_kernel<<<gg, 128, 0, stream>>>(ctx, Wo, bo, (float*)d_out, nullptr, 0);
  finalize_kernel<<<1, 32, 0, stream>>>(klb, dvb, (float*)d_out);
}
